// CausalSelfAttention_80101140071012
// MI455X (gfx1250) — compile-verified, matrix operand order checked by reading, not run on hardware
//
#include <hip/hip_runtime.h>

// ---------------------------------------------------------------------------
// CDNA5 (gfx1250, wave32) causal self-attention.
// All GEMMs: v_wmma_f32_16x16x32_bf16 (bf16 operands, f32 accumulate).
// Tile fills: global_load_async_to_lds_b128 (ASYNCcnt) + double-buffered LDS.
// 8 waves / 256 threads, 64x32 wave tile -> 64 accumulator VGPRs (no spills).
// k-loop unrolled x2 so double-buffer indices are compile-time constants.
// ---------------------------------------------------------------------------

typedef __attribute__((ext_vector_type(16))) __bf16 v16bf;
typedef __attribute__((ext_vector_type(8)))  float  v8f;
typedef __attribute__((ext_vector_type(4)))  unsigned int u32x4;

union FragAB {
  u32x4 u4[2];   // two 16B chunks
  v16bf v;       // 16 bf16 = 8 VGPRs
};

#define BM 128
#define BN 128
#define BK 32
#define LDSTR 40                 // ushorts per LDS row: 32 payload + 8 pad
#define TILE_ELEMS (BM * LDSTR)  // one LDS buffer (5120 ushorts = 10240 B)

__device__ __forceinline__ unsigned short f2bf(float f) {
  union { float f; unsigned u; } x; x.f = f;
  unsigned r = x.u + 0x7FFFu + ((x.u >> 16) & 1u);  // round-to-nearest-even
  return (unsigned short)(r >> 16);
}

__device__ __forceinline__ void async_wait0() {
  asm volatile("s_wait_asynccnt 0x0" ::: "memory");
}

// Async-copy one 128x32 bf16 tile (row-major, leading dim ld) into LDS.
// 256 threads, 2 x 16B async copies each; completion tracked by ASYNCcnt.
__device__ __forceinline__ void load_tile_async(const unsigned short* __restrict__ g,
                                                int ld, int rowBase, int kBase,
                                                unsigned short* lds) {
  const int tid = threadIdx.x;  // 0..255
#pragma unroll
  for (int i = 0; i < 2; ++i) {
    int chunk = tid + (i << 8);          // 0..511 chunks of 8 bf16
    int row = chunk >> 2;                // 0..127
    int c8  = chunk & 3;                 // 0..3
    unsigned long long ga = (unsigned long long)(uintptr_t)
        (g + (size_t)(rowBase + row) * ld + kBase + c8 * 8);
    unsigned la = (unsigned)(uintptr_t)(lds + row * LDSTR + c8 * 8);
    asm volatile("global_load_async_to_lds_b128 %0, %1, off"
                 :: "v"(la), "v"(ga) : "memory");
  }
}

// C(128x128,f32) += A(128xK) * B(128xK)^T   (B stored [N][K] row-major).
// 8 waves: wave (wid&1) -> 64 rows of M, wave (wid>>1) -> 32 cols of N.
// 8 WMMA per wave per 32-K step; double-buffered LDS, 1 barrier per step.
__device__ __forceinline__ void gemm_tile(const unsigned short* __restrict__ A, int lda,
                                          const unsigned short* __restrict__ B, int ldb,
                                          int mBase, int nBase, int kIters,
                                          unsigned short* ldsA, unsigned short* ldsB,
                                          v8f acc[4][2]) {
  const int tid  = threadIdx.x;
  const int lane = tid & 31;
  const int wid  = tid >> 5;
  const int waveM = (wid & 1) * 64;
  const int waveN = (wid >> 1) * 32;
  const int h = lane >> 4;   // 16-lane half -> K sub-block
  const int r = lane & 15;   // A row / B column within fragment

  // prologue: fill buffer 0
  load_tile_async(A, lda, mBase, 0, ldsA);
  load_tile_async(B, ldb, nBase, 0, ldsB);

#pragma unroll 2
  for (int kb = 0; kb < kIters; ++kb) {
    const int cur = kb & 1;
    async_wait0();      // this wave's pending fills (buffer `cur`) done
    __syncthreads();    // all waves' fills visible

    if (kb + 1 < kIters) {  // overlap next tile fill with this tile's WMMAs
      const int nxt = (kb + 1) & 1;
      load_tile_async(A, lda, mBase, (kb + 1) * BK, ldsA + nxt * TILE_ELEMS);
      load_tile_async(B, ldb, nBase, (kb + 1) * BK, ldsB + nxt * TILE_ELEMS);
    }

    const unsigned short* bufA = ldsA + cur * TILE_ELEMS;
    const unsigned short* bufB = ldsB + cur * TILE_ELEMS;

    FragAB aF[4], bF[2];
#pragma unroll
    for (int mi = 0; mi < 4; ++mi) {
      const unsigned short* rp = bufA + (waveM + mi * 16 + r) * LDSTR;
      aF[mi].u4[0] = *(const u32x4*)(rp + 8 * h);        // K = 8h..8h+7
      aF[mi].u4[1] = *(const u32x4*)(rp + 16 + 8 * h);   // K = 16+8h..+7
    }
#pragma unroll
    for (int ni = 0; ni < 2; ++ni) {
      const unsigned short* rp = bufB + (waveN + ni * 16 + r) * LDSTR;
      bF[ni].u4[0] = *(const u32x4*)(rp + 8 * h);
      bF[ni].u4[1] = *(const u32x4*)(rp + 16 + 8 * h);
    }
#pragma unroll
    for (int mi = 0; mi < 4; ++mi)
#pragma unroll
      for (int ni = 0; ni < 2; ++ni)
        acc[mi][ni] = __builtin_amdgcn_wmma_f32_16x16x32_bf16(
            false, aF[mi].v, false, bF[ni].v, (short)0, acc[mi][ni],
            false, false);
  }
}

#define DECL_ACC(acc)                                    \
  v8f acc[4][2];                                         \
  _Pragma("unroll") for (int mi = 0; mi < 4; ++mi)       \
  _Pragma("unroll") for (int ni = 0; ni < 2; ++ni) acc[mi][ni] = (v8f)0.0f;

// ---------------------------------------------------------------------------

__global__ void k_convert(const float* __restrict__ src,
                          unsigned short* __restrict__ dst, int n) {
  for (int i = blockIdx.x * blockDim.x + threadIdx.x; i < n;
       i += gridDim.x * blockDim.x)
    dst[i] = f2bf(src[i]);
}

// Q/K/V projection: C_z = X * W_z^T (bf16 out).  grid = (8, 64, 3), 256 thr.
__global__ void __launch_bounds__(256)
k_qkv(const unsigned short* __restrict__ A,
      const unsigned short* __restrict__ Wall,
      unsigned short* __restrict__ Call) {
  __shared__ __align__(16) unsigned short ldsA[2 * TILE_ELEMS];
  __shared__ __align__(16) unsigned short ldsB[2 * TILE_ELEMS];
  const unsigned short* B = Wall + (size_t)blockIdx.z * 1024 * 1024;
  unsigned short* C       = Call + (size_t)blockIdx.z * 8192 * 1024;
  const int mBase = blockIdx.y * BM, nBase = blockIdx.x * BN;

  DECL_ACC(acc)
  gemm_tile(A, 1024, B, 1024, mBase, nBase, 1024 / BK, ldsA, ldsB, acc);

  const int lane = threadIdx.x & 31, wid = threadIdx.x >> 5;
  const int mW = mBase + (wid & 1) * 64, nW = nBase + (wid >> 1) * 32;
  const int h = lane >> 4, n = lane & 15;
#pragma unroll
  for (int mi = 0; mi < 4; ++mi)
#pragma unroll
    for (int ni = 0; ni < 2; ++ni)
#pragma unroll
      for (int j = 0; j < 8; ++j)
        C[(size_t)(mW + mi * 16 + 8 * h + j) * 1024 + nW + ni * 16 + n] =
            f2bf(acc[mi][ni][j]);
}

// V [b][s][d] -> Vt [b][d][s]  (bf16), 32x32 tiles via LDS.  256 thr.
__global__ void k_transpose(const unsigned short* __restrict__ V,
                            unsigned short* __restrict__ Vt) {
  __shared__ unsigned short t[32][33];
  const int b = blockIdx.z;
  const int sBase = blockIdx.x * 32, dBase = blockIdx.y * 32;
  const int tx = threadIdx.x & 31, ty = threadIdx.x >> 5;  // ty 0..7
#pragma unroll
  for (int j = 0; j < 4; ++j)
    t[ty + j * 8][tx] =
        V[((size_t)b * 2048 + sBase + ty + j * 8) * 1024 + dBase + tx];
  __syncthreads();
#pragma unroll
  for (int j = 0; j < 4; ++j)
    Vt[(size_t)b * 1024 * 2048 + (size_t)(dBase + ty + j * 8) * 2048 + sBase + tx] =
        t[tx][ty + j * 8];
}

// Scores = (Q * K^T) / 32, lower-triangle tiles only.  grid = (16,16), 256 thr.
__global__ void __launch_bounds__(256)
k_scores(const unsigned short* __restrict__ Q,
         const unsigned short* __restrict__ Km,
         float* __restrict__ S) {
  const int mT = blockIdx.y, nT = blockIdx.x;
  if (nT > mT) return;  // fully masked block
  __shared__ __align__(16) unsigned short ldsA[2 * TILE_ELEMS];
  __shared__ __align__(16) unsigned short ldsB[2 * TILE_ELEMS];

  DECL_ACC(acc)
  gemm_tile(Q, 1024, Km, 1024, mT * BM, nT * BN, 1024 / BK, ldsA, ldsB, acc);

  const int lane = threadIdx.x & 31, wid = threadIdx.x >> 5;
  const int mW = mT * BM + (wid & 1) * 64, nW = nT * BN + (wid >> 1) * 32;
  const int h = lane >> 4, n = lane & 15;
#pragma unroll
  for (int mi = 0; mi < 4; ++mi)
#pragma unroll
    for (int ni = 0; ni < 2; ++ni)
#pragma unroll
      for (int j = 0; j < 8; ++j)
        S[(size_t)(mW + mi * 16 + 8 * h + j) * 2048 + nW + ni * 16 + n] =
            acc[mi][ni][j] * 0.03125f;  // 1/sqrt(1024)
}

// Row-wise causal softmax: P[q][k] = softmax(S[q][0..q]), 0 above diagonal.
__global__ void k_softmax(const float* __restrict__ S,
                          unsigned short* __restrict__ P) {
  const int q = blockIdx.x;
  const float* srow = S + (size_t)q * 2048;
  unsigned short* prow = P + (size_t)q * 2048;
  const int len = q + 1;
  const int tid = threadIdx.x;
  __shared__ float red[256];

  float vals[8];
  const int nIter = (len + 255) >> 8;
  float m = -3.0e38f;
  for (int i = 0; i < nIter; ++i) {
    int k = tid + (i << 8);
    float v = (k < len) ? srow[k] : -3.0e38f;
    vals[i] = v;
    m = fmaxf(m, v);
  }
  red[tid] = m; __syncthreads();
  for (int s = 128; s > 0; s >>= 1) {
    if (tid < s) red[tid] = fmaxf(red[tid], red[tid + s]);
    __syncthreads();
  }
  m = red[0]; __syncthreads();

  float sum = 0.0f;
  for (int i = 0; i < nIter; ++i) {
    int k = tid + (i << 8);
    float e = (k < len) ? __expf(vals[i] - m) : 0.0f;
    vals[i] = e;
    sum += e;
  }
  red[tid] = sum; __syncthreads();
  for (int s = 128; s > 0; s >>= 1) {
    if (tid < s) red[tid] += red[tid + s];
    __syncthreads();
  }
  const float inv = 1.0f / red[0];

#pragma unroll
  for (int i = 0; i < 8; ++i) {
    int k = tid + (i << 8);
    float e = (i < nIter) ? vals[i] * inv : 0.0f;
    prow[k] = f2bf(e);
  }
}

// O = P * V  (via Vt, f32 out).  K-loop bounded by causality.  grid=(8,16),256thr.
__global__ void __launch_bounds__(256)
k_pv(const unsigned short* __restrict__ P,
     const unsigned short* __restrict__ Vt,
     float* __restrict__ O) {
  __shared__ __align__(16) unsigned short ldsA[2 * TILE_ELEMS];
  __shared__ __align__(16) unsigned short ldsB[2 * TILE_ELEMS];
  const int mBase = blockIdx.y * BM;     // query rows
  const int nBase = blockIdx.x * BN;     // model-dim cols
  const int kIters = (mBase + BM) / BK;  // P is zero for k > q

  DECL_ACC(acc)
  gemm_tile(P, 2048, Vt, 2048, mBase, nBase, kIters, ldsA, ldsB, acc);

  const int lane = threadIdx.x & 31, wid = threadIdx.x >> 5;
  const int mW = mBase + (wid & 1) * 64, nW = nBase + (wid >> 1) * 32;
  const int h = lane >> 4, n = lane & 15;
#pragma unroll
  for (int mi = 0; mi < 4; ++mi)
#pragma unroll
    for (int ni = 0; ni < 2; ++ni)
#pragma unroll
      for (int j = 0; j < 8; ++j)
        O[(size_t)(mW + mi * 16 + 8 * h + j) * 1024 + nW + ni * 16 + n] =
            acc[mi][ni][j];
}

// ---------------------------------------------------------------------------

extern "C" void kernel_launch(void* const* d_in, const int* in_sizes, int n_in,
                              void* d_out, int out_size, void* d_ws, size_t ws_size,
                              hipStream_t stream) {
  (void)in_sizes; (void)n_in; (void)out_size; (void)ws_size;
  const float* X  = (const float*)d_in[0];
  const float* Wq = (const float*)d_in[1];
  const float* Wk = (const float*)d_in[2];
  const float* Wv = (const float*)d_in[3];
  float* Out = (float*)d_out;

  char* ws = (char*)d_ws;
  const size_t MB = (size_t)1 << 20;
  // Workspace layout (94 MB total):
  unsigned short* Xbf = (unsigned short*)(ws);              // 16 MB
  unsigned short* Wbf = (unsigned short*)(ws + 16 * MB);    //  6 MB (Q,K,V weights)
  unsigned short* Qbf = (unsigned short*)(ws + 22 * MB);    // 16 MB  (z=0)
  unsigned short* Kbf = (unsigned short*)(ws + 38 * MB);    // 16 MB  (z=1)
  unsigned short* Vbf = (unsigned short*)(ws + 54 * MB);    // 16 MB  (z=2)
  float*          Sc  = (float*)(ws + 54 * MB);             // reuses dead V region
  unsigned short* Vt  = (unsigned short*)(ws + 70 * MB);    // 16 MB
  unsigned short* Pm  = (unsigned short*)(ws + 86 * MB);    //  8 MB

  // 1) fp32 -> bf16 conversions
  k_convert<<<2048, 256, 0, stream>>>(X,  Xbf, 8192 * 1024);
  k_convert<<<512,  256, 0, stream>>>(Wq, Wbf,           1024 * 1024);
  k_convert<<<512,  256, 0, stream>>>(Wk, Wbf + 1048576, 1024 * 1024);
  k_convert<<<512,  256, 0, stream>>>(Wv, Wbf + 2097152, 1024 * 1024);

  // 2) Q/K/V projections (one launch, grid.z selects weight/output)
  k_qkv<<<dim3(1024 / BN, 8192 / BM, 3), 256, 0, stream>>>(Xbf, Wbf, Qbf);

  // 3) V -> Vt
  k_transpose<<<dim3(64, 32, 4), 256, 0, stream>>>(Vbf, Vt);

  // 4) per-batch attention (sequential so score/P scratch is reused)
  for (int b = 0; b < 4; ++b) {
    const unsigned short* Qb  = Qbf + (size_t)b * 2048 * 1024;
    const unsigned short* Kb  = Kbf + (size_t)b * 2048 * 1024;
    const unsigned short* Vtb = Vt  + (size_t)b * 1024 * 2048;
    float* Ob = Out + (size_t)b * 2048 * 1024;

    k_scores <<<dim3(16, 16), 256, 0, stream>>>(Qb, Kb, Sc);
    k_softmax<<<2048,         256, 0, stream>>>(Sc, Pm);
    k_pv     <<<dim3(8, 16),  256, 0, stream>>>(Pm, Vtb, Ob);
  }
}
